// Batch_Interval_Refine_75788992905572
// MI455X (gfx1250) — compile-verified, matrix operand order checked by reading, not run on hardware
//
#include <hip/hip_runtime.h>
#include <hip/hip_bf16.h>
#include <math.h>

// MI455X (gfx1250) implementation. wave32, WMMA f32<=f16 16x16x32.
// Problem is tiny (~30 GFLOP, ~10 MB working set): entirely L2-resident,
// latency-bound on the 512-step GRU recurrence. Design:
//  - recurrent weights (whh^T) resident in VGPRs as WMMA B-fragments
//  - h state in LDS (f32 master + f16 WMMA copy)
//  - per-step skip when no interval in the tile is active
//  - xp_t rows double-buffered in LDS via GLOBAL_LOAD_ASYNC_TO_LDS_B128
//    (ASYNCcnt), so the step-t+1 row streams in behind step-t compute.

typedef __attribute__((ext_vector_type(16))) _Float16 v16h;
typedef __attribute__((ext_vector_type(8)))  float    v8f;

#define Bsz  8
#define Nn   512
#define Dd   128
#define Cc   5
#define TOTI 67

__device__ __forceinline__ float sigm(float x){ return 1.0f/(1.0f+expf(-x)); }

// f16 16x32 A-fragment K index for vector element e (0..15) and lane group
// (lane>>4), per CDNA5 ISA 7.12.2. B (32x16) uses the mirrored mapping with
// N = lane&15.
__device__ __forceinline__ int kmap16(int e, int grp){
  int v = e >> 1, h = e & 1;
  int base = (v < 4) ? (2*v) : (16 + 2*(v-4));
  return base + 8*grp + h;
}

#define WMMA(a,b,c) __builtin_amdgcn_wmma_f32_16x16x32_f16(false,(a),false,(b),(short)0,(c),false,false)

// ---------------------------------------------------------------- prep ----
__global__ void k_prep_x(const float* __restrict__ emb, const float* __restrict__ tp,
                         float* __restrict__ x){
  int row = blockIdx.x;           // b*512 + n
  int d   = threadIdx.x;          // 0..127
  float t = tp[row];
  float freq = 10.0f * (float)d / 127.0f;     // linspace(0,10,128)
  x[row*Dd + d] = emb[row*Dd + d] + 0.05f * sinf(t * freq);
}

__global__ void k_prep_abn(const float* __restrict__ pred, const float* __restrict__ tp,
                           const float* __restrict__ alen,
                           float* __restrict__ abn, float* __restrict__ tq){
  int idx = blockIdx.x*blockDim.x + threadIdx.x;
  if (idx >= Bsz*Nn) return;
  int b = idx / Nn, n = idx % Nn;
  const float G[5] = {0.06136f,0.24477f,0.38774f,0.24477f,0.06136f};
  float sm[Cc];
  for (int c=0;c<Cc;c++){
    float s = 0.f;
    for (int k=0;k<5;k++){
      int nn = n + k - 2;
      if (nn >= 0 && nn < Nn) s += G[k]*pred[(b*Nn+nn)*Cc + c];
    }
    sm[c] = s;
  }
  float mx = sm[0];
  for (int c=1;c<Cc;c++) mx = fmaxf(mx, sm[c]);
  float den = 0.f, e0 = 0.f;
  for (int c=0;c<Cc;c++){ float e = expf(sm[c]-mx); den += e; if (c==0) e0 = e; }
  abn[idx] = e0/den;
  tq[idx]  = tp[idx]*alen[0];
}

// ------------------------------------------------------- x_proj GEMM ------
// out[row][g] = sum_k x[row][k]*wih[g][k] + bih[g]; M=4096, N=384, K=128.
__global__ void k_xproj(const float* __restrict__ x, const float* __restrict__ wih,
                        const float* __restrict__ bih, float* __restrict__ out){
  __shared__ _Float16 As[16][128];
  int tid = threadIdx.x;
  int row0 = blockIdx.x*16;
  for (int j = tid; j < 16*128; j += 256){
    int m = j >> 7, k = j & 127;
    As[m][k] = (_Float16)x[(row0+m)*Dd + k];
  }
  __syncthreads();
  int w = tid >> 5, lane = tid & 31;
  int ml = lane & 15, grp = lane >> 4;
  v16h a[4];
  #pragma unroll
  for (int kc=0;kc<4;kc++)
    #pragma unroll
    for (int e=0;e<16;e++)
      a[kc][e] = As[ml][kc*32 + kmap16(e,grp)];
  for (int nt=0; nt<3; nt++){
    int n0 = (w*3+nt)*16;
    v8f acc = {};
    #pragma unroll
    for (int kc=0;kc<4;kc++){
      v16h bf;
      #pragma unroll
      for (int e=0;e<16;e++){
        int k = kc*32 + kmap16(e,grp);
        bf[e] = (_Float16)wih[(n0+ml)*Dd + k];   // B(k,n)=wih[n][k]
      }
      acc = WMMA(a[kc], bf, acc);
    }
    #pragma unroll
    for (int r=0;r<8;r++){
      int mm = r + 8*grp;
      out[(row0+mm)*384 + n0 + ml] = acc[r] + bih[n0+ml];
    }
  }
}

// --------------------------------------------------------- feat GRU -------
// One block per (batch, 16-interval tile). whh^T B-fragments live in VGPRs
// for the whole 512-step scan; h kept in LDS (f32 master + f16 WMMA copy).
// xp_t rows double-buffered in LDS via async global->LDS b128 copies.
__global__ void k_gru(const float* __restrict__ xproj, const float* __restrict__ tq,
                      const float* __restrict__ intv, const float* __restrict__ whh,
                      const float* __restrict__ bih,  const float* __restrict__ bhh,
                      float* __restrict__ feat, int I, int offs, int nTiles){
  __shared__ float    hS[16][128];
  __shared__ _Float16 hH[16][128];
  __shared__ float    gh[16][384];
  __shared__ __align__(16) float xpD[2][384];
  __shared__ float    st[16], en[16];
  __shared__ int      anyM[16];
  __shared__ float    tS;

  int tid = threadIdx.x;
  int b   = blockIdx.x / nTiles;
  int i0  = (blockIdx.x % nTiles)*16;

  for (int j = tid; j < 16*128; j += 256){
    int m = j>>7, k = j&127;
    hS[m][k] = 0.f; hH[m][k] = (_Float16)0.f;
  }
  if (tid < 16){
    int i = i0 + tid;
    if (i < I){ st[tid] = intv[(b*TOTI+offs+i)*2];
                en[tid] = intv[(b*TOTI+offs+i)*2+1]; }
    else      { st[tid] = 1e30f; en[tid] = -1e30f; }
    anyM[tid] = 0;
  }

  int w = tid >> 5, lane = tid & 31;
  int ml = lane & 15, grp = lane >> 4;

  // whh^T fragments (B: K=128 x N=384), invariant across the scan.
  v16h bw[3][4];
  for (int nt=0;nt<3;nt++){
    int n0 = (w*3+nt)*16;
    #pragma unroll
    for (int kc=0;kc<4;kc++)
      #pragma unroll
      for (int e=0;e<16;e++){
        int k = kc*32 + kmap16(e,grp);
        bw[nt][kc][e] = (_Float16)whh[(n0+ml)*Dd + k];  // B(k,n)=whh[n][k]
      }
  }

  // 384 floats per row = 96 lanes x b128: waves 0..2 each issue exactly one
  // async op per step, so ASYNCcnt<=1 after an issue means the previous
  // step's row has landed (async loads complete in order).
  const float* xbase = xproj + (size_t)b*Nn*384;
  if (tid < 96){
    unsigned ldsa = (unsigned)(size_t)&xpD[0][tid*4];
    const float* g = xbase + tid*4;
    asm volatile("global_load_async_to_lds_b128 %0, %1, off"
                 :: "v"(ldsa), "v"(g) : "memory");
  }
  __syncthreads();

  for (int t=0;t<Nn;t++){
    int cur = t & 1;
    if (t+1 < Nn){
      if (tid < 96){
        unsigned ldsa = (unsigned)(size_t)&xpD[cur^1][tid*4];
        const float* g = xbase + (size_t)(t+1)*384 + tid*4;
        asm volatile("global_load_async_to_lds_b128 %0, %1, off"
                     :: "v"(ldsa), "v"(g) : "memory");
      }
      asm volatile("s_wait_asynccnt 0x1" ::: "memory");
    } else {
      asm volatile("s_wait_asynccnt 0x0" ::: "memory");
    }
    if (tid==0) tS = tq[b*Nn+t];
    __syncthreads();
    const float* xp = xpD[cur];
    float tv = tS;
    // skip steps where no interval in this tile is active
    int act = __syncthreads_or((tid < 16) && (tv >= st[tid]) && (tv <= en[tid]));
    if (!act) continue;

    v16h a[4];
    #pragma unroll
    for (int kc=0;kc<4;kc++)
      #pragma unroll
      for (int e=0;e<16;e++)
        a[kc][e] = hH[ml][kc*32 + kmap16(e,grp)];
    for (int nt=0;nt<3;nt++){
      v8f acc = {};
      #pragma unroll
      for (int kc=0;kc<4;kc++)
        acc = WMMA(a[kc], bw[nt][kc], acc);
      int n0 = (w*3+nt)*16;
      #pragma unroll
      for (int r=0;r<8;r++) gh[r+8*grp][n0+ml] = acc[r];
    }
    __syncthreads();
    for (int u=tid;u<2048;u+=256){
      int m = u>>7, j = u&127;
      if (tv >= st[m] && tv <= en[m]){
        float r  = sigm (xp[j]       + gh[m][j]       + bhh[j]);
        float z  = sigm (xp[128+j]   + gh[m][128+j]   + bhh[128+j]);
        float nn = tanhf(xp[256+j]   + r*(gh[m][256+j]+ bhh[256+j]));
        float hn = (1.f - z)*nn + z*hS[m][j];
        hS[m][j] = hn; hH[m][j] = (_Float16)hn;
        if (j==0) anyM[m] = 1;
      }
    }
    __syncthreads();
  }
  // finalize: empty-mask rows get h_empty computed from biases only
  for (int u=tid;u<2048;u+=256){
    int m = u>>7, j = u&127;
    int i = i0+m;
    if (i < I){
      float v;
      if (anyM[m]) v = hS[m][j];
      else {
        float r0 = sigm(bih[j]     + bhh[j]);
        float z0 = sigm(bih[128+j] + bhh[128+j]);
        float n0 = tanhf(bih[256+j] + r0*bhh[256+j]);
        v = (1.f - z0)*n0;
      }
      feat[(b*TOTI+offs+i)*Dd + j] = v;
    }
  }
}

// ----------------------------------------------------------- abn GRU ------
__global__ void k_abn(const float* __restrict__ abn, const float* __restrict__ tq,
                      const float* __restrict__ intv,
                      const float* __restrict__ wihA, const float* __restrict__ bihA,
                      const float* __restrict__ whhA, const float* __restrict__ bhhA,
                      float* __restrict__ labn, int I, int offs){
  int idx = blockIdx.x*blockDim.x + threadIdx.x;
  if (idx >= Bsz*I) return;
  int b = idx / I, i = idx % I;
  float s0 = intv[(b*TOTI+offs+i)*2], e0 = intv[(b*TOTI+offs+i)*2+1];
  float wi0=wihA[0],wi1=wihA[1],wi2=wihA[2];
  float bi0=bihA[0],bi1=bihA[1],bi2=bihA[2];
  float wh0=whhA[0],wh1=whhA[1],wh2=whhA[2];
  float bh0=bhhA[0],bh1=bhhA[1],bh2=bhhA[2];
  float h = 0.f; bool any = false;
  for (int t=0;t<Nn;t++){
    float tv = tq[b*Nn+t];
    if (tv >= s0 && tv <= e0){
      float a  = abn[b*Nn+t];
      float r  = sigm (a*wi0+bi0 + h*wh0+bh0);
      float z  = sigm (a*wi1+bi1 + h*wh1+bh1);
      float nn = tanhf(a*wi2+bi2 + r*(h*wh2+bh2));
      h = (1.f - z)*nn + z*h; any = true;
    }
  }
  if (!any){
    float r0 = sigm(bi0+bh0), z0 = sigm(bi1+bh1);
    float n0 = tanhf(bi2 + r0*bh2);
    h = (1.f - z0)*n0;
  }
  labn[b*TOTI+offs+i] = h;
}

// -------------------------------------------------------- fused heads -----
// 131 -> 256 -> 256 -> 37 MLP on 16-row tiles, activations staged as f16 in
// LDS; then per-row bin softmax + weighted sum + clip, and output assembly.
__global__ void k_head(const float* __restrict__ feat, const float* __restrict__ labn,
                       const float* __restrict__ intv, const float* __restrict__ alenp,
                       const float* __restrict__ w1, const float* __restrict__ b1,
                       const float* __restrict__ w2, const float* __restrict__ b2,
                       const float* __restrict__ w3, const float* __restrict__ b3,
                       const float* __restrict__ wp, float* __restrict__ outp,
                       int I, int offs){
  __shared__ _Float16 buf[16][256];
  __shared__ float    hO[16][48];
  __shared__ float    stS[16], enS[16];
  __shared__ int      validS[16];
  int tid = threadIdx.x;
  float alen = alenp[0];
  int r0 = blockIdx.x*16;
  if (tid < 16){
    int ri = r0 + tid;
    int valid = ri < Bsz*I;
    validS[tid] = valid;
    if (valid){
      int b = ri / I, i = ri % I;
      stS[tid] = intv[(b*TOTI+offs+i)*2];
      enS[tid] = intv[(b*TOTI+offs+i)*2+1];
    } else { stS[tid] = 0.f; enS[tid] = 0.f; }
  }
  __syncthreads();
  // stage shared = [feat(128) | labn | center/alen | width/alen | zeros]
  for (int j=tid;j<16*256;j+=256){
    int m = j>>8, col = j&255;
    int ri = r0 + m;
    _Float16 v = (_Float16)0.f;
    if (validS[m]){
      int b = ri / I, i = ri % I;
      int g = b*TOTI+offs+i;
      if      (col < 128)  v = (_Float16)feat[g*Dd+col];
      else if (col == 128) v = (_Float16)labn[g];
      else if (col == 129) v = (_Float16)(((stS[m]+enS[m])*0.5f)/alen);
      else if (col == 130) v = (_Float16)((enS[m]-stS[m])/alen);
    }
    buf[m][col] = v;
  }
  __syncthreads();
  int w = tid>>5, lane = tid&31;
  int ml = lane&15, grp = lane>>4;

  // layer1: K=131 (padded to 160), N=256
  v8f acc1[2];
  for (int nt=0;nt<2;nt++){
    int n0 = (w*2+nt)*16;
    v8f acc = {};
    for (int kc=0;kc<5;kc++){
      v16h a, bf;
      #pragma unroll
      for (int e=0;e<16;e++){
        int k = kc*32 + kmap16(e,grp);
        a[e]  = buf[ml][k];
        bf[e] = (k < 131) ? (_Float16)w1[(n0+ml)*131 + k] : (_Float16)0.f;
      }
      acc = WMMA(a, bf, acc);
    }
    acc1[nt] = acc;
  }
  __syncthreads();
  for (int nt=0;nt<2;nt++){
    int n0 = (w*2+nt)*16;
    #pragma unroll
    for (int r=0;r<8;r++){
      int mm = r + 8*grp;
      buf[mm][n0+ml] = (_Float16)fmaxf(acc1[nt][r] + b1[n0+ml], 0.f);
    }
  }
  __syncthreads();
  // layer2: K=256, N=256
  v8f acc2[2];
  for (int nt=0;nt<2;nt++){
    int n0 = (w*2+nt)*16;
    v8f acc = {};
    for (int kc=0;kc<8;kc++){
      v16h a, bf;
      #pragma unroll
      for (int e=0;e<16;e++){
        int k = kc*32 + kmap16(e,grp);
        a[e]  = buf[ml][k];
        bf[e] = (_Float16)w2[(n0+ml)*256 + k];
      }
      acc = WMMA(a, bf, acc);
    }
    acc2[nt] = acc;
  }
  __syncthreads();
  for (int nt=0;nt<2;nt++){
    int n0 = (w*2+nt)*16;
    #pragma unroll
    for (int r=0;r<8;r++){
      int mm = r + 8*grp;
      buf[mm][n0+ml] = (_Float16)fmaxf(acc2[nt][r] + b2[n0+ml], 0.f);
    }
  }
  __syncthreads();
  // layer3: K=256, N=37 (padded to 48)
  if (w < 3){
    int n0 = w*16;
    v8f acc = {};
    for (int kc=0;kc<8;kc++){
      v16h a, bf;
      #pragma unroll
      for (int e=0;e<16;e++){
        int k = kc*32 + kmap16(e,grp);
        a[e]  = buf[ml][k];
        bf[e] = (n0+ml < 37) ? (_Float16)w3[(n0+ml)*256 + k] : (_Float16)0.f;
      }
      acc = WMMA(a, bf, acc);
    }
    #pragma unroll
    for (int r=0;r<8;r++){
      int mm = r + 8*grp;
      hO[mm][n0+ml] = acc[r] + ((n0+ml < 37) ? b3[n0+ml] : 0.f);
    }
  }
  __syncthreads();
  if (tid < 16 && validS[tid]){
    int ri = r0 + tid;
    int b = ri / I, i = ri % I, g = b*TOTI+offs+i;
    const float* h = hO[tid];
    float mx1 = h[0]; for (int j=1;j<16;j++)  mx1 = fmaxf(mx1, h[j]);
    float d1=0.f, s1=0.f;
    for (int j=0;j<16;j++){ float e = expf(h[j]-mx1); d1 += e; s1 += e*wp[j]; }
    float mx2 = h[16]; for (int j=17;j<32;j++) mx2 = fmaxf(mx2, h[j]);
    float d2=0.f, s2=0.f;
    for (int j=16;j<32;j++){ float e = expf(h[j]-mx2); d2 += e; s2 += e*wp[j-16]; }
    float ns = fminf(fmaxf(stS[tid] + s1/d1, 0.f), alen);
    float ne = fminf(fmaxf(enS[tid] + s2/d2, 0.f), alen);
    outp[g*2]   = ns;
    outp[g*2+1] = ne;
    outp[Bsz*TOTI*2 + g] = h[32];
    for (int c=0;c<4;c++) outp[Bsz*TOTI*2 + Bsz*TOTI + g*4 + c] = h[33+c];
  }
}

// ------------------------------------------------------------- launch -----
extern "C" void kernel_launch(void* const* d_in, const int* in_sizes, int n_in,
                              void* d_out, int out_size, void* d_ws, size_t ws_size,
                              hipStream_t stream){
  (void)in_sizes; (void)n_in; (void)out_size; (void)ws_size;
  const float* emb  = (const float*)d_in[0];
  const float* tp   = (const float*)d_in[1];
  const float* pred = (const float*)d_in[2];
  const float* intv = (const float*)d_in[3];
  const float* alen = (const float*)d_in[4];
  // params leaves in jax pytree order (sorted dict keys):
  //   abn_gru[s]:  {bhh,bih,whh,wih} at 5+4s
  //   feat_gru[s]: {bhh,bih,whh,wih} at 17+4s
  //   heads[s]:    {b,w} x 3 layers at 29+6s
  //   weight_params[s]: 47+s ; n1..n3 at 50..52 (unused)
  float* ws    = (float*)d_ws;
  float* xbuf  = ws;                           // 8*512*128
  float* abn   = xbuf  + Bsz*Nn*Dd;            // 8*512
  float* tq    = abn   + Bsz*Nn;               // 8*512
  float* xproj = tq    + Bsz*Nn;               // 8*512*384 (reused per scale)
  float* feat  = xproj + Bsz*Nn*384;           // 8*67*128
  float* labn  = feat  + Bsz*TOTI*Dd;          // 8*67
  float* outp  = (float*)d_out;

  k_prep_x  <<<Bsz*Nn, 128, 0, stream>>>(emb, tp, xbuf);
  k_prep_abn<<<(Bsz*Nn+255)/256, 256, 0, stream>>>(pred, tp, alen, abn, tq);

  const int NPSs[3] = {39,19,9}, OFFS[3] = {0,39,58};
  for (int s=0;s<3;s++){
    const float* abhh=(const float*)d_in[5+4*s+0];
    const float* abih=(const float*)d_in[5+4*s+1];
    const float* awhh=(const float*)d_in[5+4*s+2];
    const float* awih=(const float*)d_in[5+4*s+3];
    const float* fbhh=(const float*)d_in[17+4*s+0];
    const float* fbih=(const float*)d_in[17+4*s+1];
    const float* fwhh=(const float*)d_in[17+4*s+2];
    const float* fwih=(const float*)d_in[17+4*s+3];
    int I = NPSs[s], offs = OFFS[s];
    int nT = (I+15)/16;
    k_xproj<<<(Bsz*Nn)/16, 256, 0, stream>>>(xbuf, fwih, fbih, xproj);
    k_gru  <<<Bsz*nT, 256, 0, stream>>>(xproj, tq, intv, fwhh, fbih, fbhh, feat, I, offs, nT);
    k_abn  <<<(Bsz*I+255)/256, 256, 0, stream>>>(abn, tq, intv, awih, abih, awhh, abhh, labn, I, offs);
  }
  for (int s=0;s<3;s++){
    const float* b1=(const float*)d_in[29+6*s+0];
    const float* w1=(const float*)d_in[29+6*s+1];
    const float* b2=(const float*)d_in[29+6*s+2];
    const float* w2=(const float*)d_in[29+6*s+3];
    const float* b3=(const float*)d_in[29+6*s+4];
    const float* w3=(const float*)d_in[29+6*s+5];
    const float* wp=(const float*)d_in[47+s];
    int I = NPSs[s], offs = OFFS[s];
    int nT = (Bsz*I+15)/16;
    k_head<<<nT, 256, 0, stream>>>(feat, labn, intv, alen, w1,b1,w2,b2,w3,b3, wp, outp, I, offs);
  }
}
